// ESMGearNet_56324201120046
// MI455X (gfx1250) — matrix operand order repaired and checked by reading, not compiled, not run on hardware
//
// ESM-GearNet forward for MI455X (gfx1250), compile-only target.
//
// Memory-bound (~25-30 GB scatter/gather + aggregates vs ~1.6 TFLOP matmul ->
// HBM floor ~1.2ms @ 23.3 TB/s). All GEMMs route through wave32 WMMA
// v_wmma_f32_16x16x32_bf16 (f32 accumulate) with fused bias/BN/ReLU/residual
// epilogues. Relational aggregation uses global f32 atomics; the [E, 8*512]
// line-graph aggregate is computed in 8 dst-chunks of [8192, 4096] (chunking is
// ~free: each line-graph edge lands in exactly one chunk).
//
// This revision removes per-element masking from the GEMM stage-in (split
// main/tail K loop), uses global_load_b128 (float4) + ds_store_b64 staging on
// the common aligned path, and float4 gathers in the scatter kernels.
//
// Numerical note: GEMM operands are cast f32->bf16 (RNE); accumulate is f32.

#include <hip/hip_runtime.h>
#include <hip/hip_bf16.h>
#include <stdint.h>

#define NN    8192
#define NE    65536
#define NLG   262144
#define NGR   16
#define NRL   7
#define NAB_R 8
#define OUTD  3072
#define EPSV  1e-5f

#define BM 64
#define BN 64
#define BK 32
#define LDSROW 40   // ushorts per LDS row: 32 data + 8 pad = 80B (16B-aligned, conflict-free)

typedef __bf16 bf16_t;
typedef bf16_t v16bf __attribute__((ext_vector_type(16)));
typedef float  v8f   __attribute__((ext_vector_type(8)));
typedef unsigned int uint4v __attribute__((ext_vector_type(4)));
typedef unsigned short u16;

union Frag16 { v16bf v; uint4v q[2]; };

__device__ __forceinline__ u16 f2bf(float f) {
  unsigned int u = __float_as_uint(f);
  u += 0x7FFFu + ((u >> 16) & 1u);   // round-to-nearest-even
  return (u16)(u >> 16);
}

// -------------------------------------------------------------------------
// Relational scatter-aggregate:
//   agg[(dst-lo)*nrel + rel][:] += x[src or e][:] * w[e]   for dst in [lo,hi)
// One 128-thread block per edge; float4 gathers + f32 global atomics.
// -------------------------------------------------------------------------
__global__ __launch_bounds__(128)
void scatter_rel_agg(const float* __restrict__ x, int ldx,
                     const int* __restrict__ src, const int* __restrict__ dst,
                     const int* __restrict__ rel, const float* __restrict__ w,
                     int nE, int d, int nrel, int lo, int hi,
                     float* __restrict__ agg) {
  int e = blockIdx.x;
  if (e >= nE) return;
  int de = dst[e];
  if (de < lo || de >= hi) return;
  int s = src ? src[e] : e;         // src==NULL -> per-edge features (upd path)
  int r = rel[e];
  float we = w[e];
  const float* xr = x + (size_t)s * ldx;
  float* ar = agg + ((size_t)(de - lo) * nrel + r) * d;
  if (((d | ldx) & 3) == 0) {       // common path: d=512, row 16B-aligned
    const float4* x4 = (const float4*)xr;
    int d4 = d >> 2;
    for (int i = threadIdx.x; i < d4; i += blockDim.x) {
      float4 v = x4[i];
      float* a = ar + (i << 2);
      atomicAdd(a + 0, v.x * we);
      atomicAdd(a + 1, v.y * we);
      atomicAdd(a + 2, v.z * we);
      atomicAdd(a + 3, v.w * we);
    }
  } else {                          // layer-0 dims (21 / 59)
    for (int i = threadIdx.x; i < d; i += blockDim.x)
      atomicAdd(&ar[i], xr[i] * we);
  }
}

// -------------------------------------------------------------------------
// Fused WMMA GEMM:  acc = A1@W1 (+ A2@W2)   A:[M,K] f32, W:[K,Nd] f32
// mode 0 (conv):  C = relu((acc + b1 + b2) * g*rsqrt(1+eps) + be)
// mode 1 (upd):   C = (relu(acc + b1) + res1 [+ res2]) * g*rsqrt(1+eps) + be
// mode 2 (plain): C = acc
// Block: 256 threads (8 waves); block tile 64x64; wave tile 16x32; K step 32.
// M must be a multiple of 64 (true for all calls: 8192 / 65536-chunked).
// -------------------------------------------------------------------------
__global__ __launch_bounds__(256)
void wmma_fused_gemm(const float* __restrict__ A1, int lda1, int K1,
                     const float* __restrict__ W1,
                     const float* __restrict__ A2, int lda2, int K2,
                     const float* __restrict__ W2,
                     int M, int Nd,
                     const float* __restrict__ bias1,
                     const float* __restrict__ bias2,
                     const float* __restrict__ g,
                     const float* __restrict__ be,
                     const float* __restrict__ res1, int ldr1,
                     const float* __restrict__ res2, int ldr2,
                     float* __restrict__ C, int ldc, int mode) {
  __shared__ __align__(16) u16 As[BM * LDSROW];
  __shared__ __align__(16) u16 Bs[BN * LDSROW];

  const int tid  = threadIdx.x;
  const int wave = tid >> 5;
  const int lane = tid & 31;
  const int half = lane >> 4;          // 0: lanes 0-15, 1: lanes 16-31
  const int l16  = lane & 15;
  const int wm   = (wave >> 1) * 16;   // wave row offset in block tile
  const int wn   = (wave & 1) * 32;    // wave col offset in block tile
  const int m0   = blockIdx.x * BM;
  const int n0   = blockIdx.y * BN;

  v8f acc0 = {0.f, 0.f, 0.f, 0.f, 0.f, 0.f, 0.f, 0.f};
  v8f acc1 = {0.f, 0.f, 0.f, 0.f, 0.f, 0.f, 0.f, 0.f};

  // Fragment loads (contiguous k = 16*half + s for both A and B) + 2 WMMAs.
  auto doWmma = [&]() {
    __syncthreads();
    Frag16 fa, fb0, fb1;
    const int ar = (wm + l16) * LDSROW;
    fa.q[0] = *(const uint4v*)(As + ar + half * 16);
    fa.q[1] = *(const uint4v*)(As + ar + half * 16 + 8);
    const int br0 = (wn + l16) * LDSROW;
    fb0.q[0] = *(const uint4v*)(Bs + br0 + half * 16);
    fb0.q[1] = *(const uint4v*)(Bs + br0 + half * 16 + 8);
    const int br1 = (wn + 16 + l16) * LDSROW;
    fb1.q[0] = *(const uint4v*)(Bs + br1 + half * 16);
    fb1.q[1] = *(const uint4v*)(Bs + br1 + half * 16 + 8);
    acc0 = __builtin_amdgcn_wmma_f32_16x16x32_bf16(false, fa.v, false, fb0.v,
                                                   (short)0, acc0, false, false);
    acc1 = __builtin_amdgcn_wmma_f32_16x16x32_bf16(false, fa.v, false, fb1.v,
                                                   (short)0, acc1, false, false);
    __syncthreads();
  };

  auto mm = [&](const float* A, int lda, int K, const float* W) {
    const int  kmain = K & ~(BK - 1);
    const bool vecA  = ((lda & 3) == 0);
    const bool vecB  = ((Nd & 3) == 0) && (n0 + BN <= Nd);
    int kk = 0;
    for (; kk < kmain; kk += BK) {     // unmasked main loop
      if (vecA) {                      // b128 loads, b64 LDS stores
#pragma unroll
        for (int j = 0; j < 2; ++j) {
          int idx = j * 256 + tid;     // 512 float4 slots
          int r = idx >> 3, k4 = (idx & 7) << 2;
          const float4 v = *(const float4*)(A + (size_t)(m0 + r) * lda + (kk + k4));
          uint2 p;
          p.x = ((unsigned)f2bf(v.y) << 16) | (unsigned)f2bf(v.x);
          p.y = ((unsigned)f2bf(v.w) << 16) | (unsigned)f2bf(v.z);
          *(uint2*)(As + r * LDSROW + k4) = p;
        }
      } else {                         // unaligned rows (lda = 21/59/147/...)
#pragma unroll
        for (int j = 0; j < 8; ++j) {
          int idx = j * 256 + tid;
          int r = idx >> 5, k = idx & 31;
          As[r * LDSROW + k] = f2bf(A[(size_t)(m0 + r) * lda + (kk + k)]);
        }
      }
      if (vecB) {                      // b128 loads, transposed b16 LDS stores
#pragma unroll
        for (int j = 0; j < 2; ++j) {
          int idx = j * 256 + tid;
          int k = idx >> 4, n4 = (idx & 15) << 2;
          const float4 v = *(const float4*)(W + (size_t)(kk + k) * Nd + (n0 + n4));
          Bs[(n4 + 0) * LDSROW + k] = f2bf(v.x);
          Bs[(n4 + 1) * LDSROW + k] = f2bf(v.y);
          Bs[(n4 + 2) * LDSROW + k] = f2bf(v.z);
          Bs[(n4 + 3) * LDSROW + k] = f2bf(v.w);
        }
      } else {                         // narrow output tile (Nd = 21)
#pragma unroll
        for (int j = 0; j < 8; ++j) {
          int idx = j * 256 + tid;
          int k = idx >> 6, n = idx & 63;
          int ng = n0 + n;
          float v = (ng < Nd) ? W[(size_t)(kk + k) * Nd + ng] : 0.f;
          Bs[n * LDSROW + k] = f2bf(v);
        }
      }
      doWmma();
    }
    if (kk < K) {                      // masked K tail (layer-0 shapes only)
#pragma unroll
      for (int j = 0; j < 8; ++j) {
        int idx = j * 256 + tid;
        int r = idx >> 5, k = idx & 31;
        int kg = kk + k;
        float v = (kg < K) ? A[(size_t)(m0 + r) * lda + kg] : 0.f;
        As[r * LDSROW + k] = f2bf(v);
      }
#pragma unroll
      for (int j = 0; j < 8; ++j) {
        int idx = j * 256 + tid;
        int k = idx >> 6, n = idx & 63;
        int kg = kk + k, ng = n0 + n;
        float v = (kg < K && ng < Nd) ? W[(size_t)kg * Nd + ng] : 0.f;
        Bs[n * LDSROW + k] = f2bf(v);
      }
      doWmma();
    }
  };

  mm(A1, lda1, K1, W1);
  if (A2) mm(A2, lda2, K2, W2);

  const float inv = rsqrtf(1.f + EPSV);
#pragma unroll
  for (int t = 0; t < 2; ++t) {
    v8f acc = t ? acc1 : acc0;
    int col = n0 + wn + t * 16 + l16;
    if (col >= Nd) continue;
#pragma unroll
    for (int i = 0; i < 8; ++i) {
      int row = m0 + wm + i + half * 8;   // C/D layout: VGPR i -> M=i (lo) / i+8 (hi)
      float v = acc[i];
      if (mode == 0) {
        v += bias1[col] + bias2[col];
        v = v * (g[col] * inv) + be[col];
        v = fmaxf(v, 0.f);
      } else if (mode == 1) {
        v = fmaxf(v + bias1[col], 0.f);
        v += res1[(size_t)row * ldr1 + col];
        if (res2) v += res2[(size_t)row * ldr2 + col];
        v = v * (g[col] * inv) + be[col];
      }
      C[(size_t)row * ldc + col] = v;
    }
  }
}

// -------------------------------------------------------------------------
// Graph sum readout: out[g][c] = sum_{n: n2g[n]==g} hid[n][c]
// node2graph staged in LDS (32 KB of the 320 KB WGP pool); no atomics.
// -------------------------------------------------------------------------
__global__ __launch_bounds__(256)
void graph_readout(const float* __restrict__ hid, const int* __restrict__ n2g,
                   float* __restrict__ out) {
  __shared__ int sg[NN];
  int gidx = blockIdx.x;
  int c = blockIdx.y * 256 + threadIdx.x;
  for (int i = threadIdx.x; i < NN; i += blockDim.x) sg[i] = n2g[i];
  __syncthreads();
  float acc = 0.f;
  for (int n = 0; n < NN; ++n)
    if (sg[n] == gidx) acc += hid[(size_t)n * OUTD + c];
  out[(size_t)gidx * OUTD + c] = acc;
}

// -------------------------------------------------------------------------
extern "C" void kernel_launch(void* const* d_in, const int* in_sizes, int n_in,
                              void* d_out, int out_size, void* d_ws, size_t ws_size,
                              hipStream_t stream) {
  if (n_in < 96) return;

  const float* node_feat = (const float*)d_in[0];
  const int*   e_src  = (const int*)d_in[1];
  const int*   e_dst  = (const int*)d_in[2];
  const int*   e_rel  = (const int*)d_in[3];
  const float* e_w    = (const float*)d_in[4];
  const float* edge_feat = (const float*)d_in[5];
  const int*   lg_src = (const int*)d_in[6];
  const int*   lg_dst = (const int*)d_in[7];
  const int*   lg_rel = (const int*)d_in[8];
  const float* lg_w   = (const float*)d_in[9];
  const int*   n2g    = (const int*)d_in[10];

  // Param flattening order: detect insertion-order vs sorted-key dict flatten.
  // node layer dict entry [base+1]: insertion -> b_loop (512), sorted -> W_rel (75264)
  const int base = 11;
  const bool skeys = in_sizes[base + 1] > 4096;
  int oWl, oBl, oWr, oBr, oG1, oE1, oG2, oE2;   // node layer offsets (8 arrays)
  int eWl, eBl, eWr, eBr, eG1, eE1;             // edge layer offsets (6 arrays)
  if (!skeys) {
    oWl = 0; oBl = 1; oWr = 2; oBr = 3; oG1 = 4; oE1 = 5; oG2 = 6; oE2 = 7;
    eWl = 0; eBl = 1; eWr = 2; eBr = 3; eG1 = 4; eE1 = 5;
  } else {  // sorted: W_loop, W_rel, b_loop, b_rel, be1, (be2,) g1, (g2)
    oWl = 0; oWr = 1; oBl = 2; oBr = 3; oE1 = 4; oE2 = 5; oG1 = 6; oG2 = 7;
    eWl = 0; eWr = 1; eBl = 2; eBr = 3; eE1 = 4; eG1 = 5;
  }
  const float* proj_W = (const float*)d_in[base + 48 + 36];   // index 95

  // Workspace carve-up (~496 MiB)
  float* agg  = (float*)d_ws;                     // 8192*4096            = 33554432 f
  float* heA  = agg  + (size_t)33554432;          // edge hidden ping     = 33554432 f
  float* heB  = heA  + (size_t)33554432;          // edge hidden pong     = 33554432 f
  float* htmp = heB  + (size_t)33554432;          // node conv hidden     =  4194304 f
  float* hcat = htmp + (size_t)4194304;           // concat hiddens       = 25165824 f
  size_t need = ((size_t)33554432 * 3 + 4194304 + 25165824) * sizeof(float);
  if (ws_size < need) return;

  static const int NDIM[7] = {21, 512, 512, 512, 512, 512, 512};
  static const int EDIM[7] = {59, 21, 512, 512, 512, 512, 512};

  const float* hn = node_feat; int ldn = 21;      // current node features
  const float* he = edge_feat; int lde = 59;      // current edge features
  float* heOut = heA;

  for (int i = 0; i < 6; ++i) {
    const int nbi = base + i * 8;
    const int ebi = base + 48 + i * 6;
    const float* nWloop = (const float*)d_in[nbi + oWl];
    const float* nbloop = (const float*)d_in[nbi + oBl];
    const float* nWrel  = (const float*)d_in[nbi + oWr];
    const float* nbrel  = (const float*)d_in[nbi + oBr];
    const float* ng1    = (const float*)d_in[nbi + oG1];
    const float* nbe1   = (const float*)d_in[nbi + oE1];
    const float* ng2    = (const float*)d_in[nbi + oG2];
    const float* nbe2   = (const float*)d_in[nbi + oE2];
    const float* eWloop = (const float*)d_in[ebi + eWl];
    const float* ebloop = (const float*)d_in[ebi + eBl];
    const float* eWrel  = (const float*)d_in[ebi + eWr];
    const float* ebrel  = (const float*)d_in[ebi + eBr];
    const float* eg1    = (const float*)d_in[ebi + eG1];
    const float* ebe1   = (const float*)d_in[ebi + eE1];

    const int dInE = EDIM[i], dOutE = EDIM[i + 1];
    const int dInN = NDIM[i];                 // == dOutE of this layer's edge conv

    // ---- 1) line-graph relational conv over E edge-nodes, 8 dst-chunks ----
    for (int c = 0; c < 8; ++c) {
      int lo = c * 8192;
      hipMemsetAsync(agg, 0, (size_t)8192 * NAB_R * dInE * sizeof(float), stream);
      scatter_rel_agg<<<NLG, 128, 0, stream>>>(he, lde, lg_src, lg_dst, lg_rel, lg_w,
                                               NLG, dInE, NAB_R, lo, lo + 8192, agg);
      dim3 gE(8192 / BM, (dOutE + BN - 1) / BN);
      wmma_fused_gemm<<<gE, 256, 0, stream>>>(
          agg, NAB_R * dInE, NAB_R * dInE, eWrel,
          he + (size_t)lo * lde, lde, dInE, eWloop,
          8192, dOutE,
          ebrel, ebloop, eg1, ebe1,
          nullptr, 0, nullptr, 0,
          heOut + (size_t)lo * dOutE, dOutE, 0);
    }

    // ---- 2) node relational conv ----
    hipMemsetAsync(agg, 0, (size_t)NN * NRL * dInN * sizeof(float), stream);
    scatter_rel_agg<<<NE, 128, 0, stream>>>(hn, ldn, e_src, e_dst, e_rel, e_w,
                                            NE, dInN, NRL, 0, NN, agg);
    dim3 gN(NN / BM, 512 / BN);
    wmma_fused_gemm<<<gN, 256, 0, stream>>>(
        agg, NRL * dInN, NRL * dInN, nWrel,
        hn, ldn, dInN, nWloop,
        NN, 512,
        nbrel, nbloop, ng1, nbe1,
        nullptr, 0, nullptr, 0,
        htmp, 512, 0);

    // ---- 3) edge->node update: scatter per-edge hidden into (dst, rel) ----
    hipMemsetAsync(agg, 0, (size_t)NN * NRL * dInN * sizeof(float), stream);
    scatter_rel_agg<<<NE, 128, 0, stream>>>(heOut, dOutE, nullptr, e_dst, e_rel, e_w,
                                            NE, dInN, NRL, 0, NN, agg);

    // ---- 4) upd GEMM + relu + hidden + shortcut + GearNet BN -> hcat slice ----
    float* hout = hcat + (size_t)i * 512;
    const float* r2 = (i > 0) ? (hcat + (size_t)(i - 1) * 512) : nullptr;
    wmma_fused_gemm<<<gN, 256, 0, stream>>>(
        agg, NRL * dInN, NRL * dInN, nWrel,
        nullptr, 0, 0, nullptr,
        NN, 512,
        nbrel, nullptr, ng2, nbe2,
        htmp, 512, r2, OUTD,
        hout, OUTD, 1);

    hn = hout; ldn = OUTD;
    he = heOut; lde = dOutE;
    heOut = (heOut == heA) ? heB : heA;
  }

  // ---- output projection: [8192,3072] @ [3072,256] ----
  dim3 gP(NN / BM, 256 / BN);
  wmma_fused_gemm<<<gP, 256, 0, stream>>>(
      hcat, OUTD, OUTD, proj_W,
      nullptr, 0, 0, nullptr,
      NN, 256,
      nullptr, nullptr, nullptr, nullptr,
      nullptr, 0, nullptr, 0,
      (float*)d_out, 256, 2);

  // ---- graph sum readout -> second output region ----
  dim3 gR(NGR, OUTD / 256);
  graph_readout<<<gR, 256, 0, stream>>>(hcat, n2g, (float*)d_out + (size_t)NN * 256);
}
